// GeneratorBlock_8452495639106
// MI455X (gfx1250) — matrix operand order repaired and checked by reading, not compiled, not run on hardware
//
#include <hip/hip_runtime.h>
#include <hip/hip_bf16.h>
#include <math.h>

typedef __attribute__((ext_vector_type(16))) _Float16 v16h;
typedef __attribute__((ext_vector_type(8)))  float    v8f;

#define LEAK 0.2f
#define EPS_DEMOD 1e-8f
#define BATCH 8
#define CH 128          // C_IN == C_OUT == 128
#define DLAT 512
#define RES 32
#define VOX (RES*RES*RES)   // 32768
#define TAPS 27
#define KC_PER_TAP 4        // 128 channels / 32
#define HALO 6              // (4 + 2) per dim
#define HPOS (HALO*HALO*HALO) // 216

__device__ __forceinline__ float lrelu(float v) { return v >= 0.0f ? v : LEAK * v; }

// ---------------------------------------------------------------------------
// K0: styles  s[b][o] = sum_k w_lat[b][k] * W[o][k] * c_lin + bias[o]
// grid: 3 sets * 8 batches blocks, 128 threads
// ---------------------------------------------------------------------------
__global__ void gen_styles_kernel(const float* __restrict__ w_lat,
                                  const float* __restrict__ W1, const float* __restrict__ b1,
                                  const float* __restrict__ W2, const float* __restrict__ b2,
                                  const float* __restrict__ Wv, const float* __restrict__ bv,
                                  float* __restrict__ s_out /* [3][B][CH] */) {
    const int set = blockIdx.x / BATCH;
    const int b   = blockIdx.x % BATCH;
    const int o   = threadIdx.x;                 // 0..127
    const float c_lin = 0.04419417382415922f;    // 1/sqrt(512)
    const float* W  = (set == 0) ? W1 : (set == 1) ? W2 : Wv;
    const float* bb = (set == 0) ? b1 : (set == 1) ? b2 : bv;
    const float* wl = w_lat + (size_t)b * DLAT;
    const float* wr = W + (size_t)o * DLAT;
    float acc = 0.0f;
    for (int k = 0; k < DLAT; ++k) acc += wl[k] * wr[k];
    s_out[((size_t)set * BATCH + b) * CH + o] = acc * c_lin + bb[o];
}

// ---------------------------------------------------------------------------
// A-operand swizzle for v_wmma_f32_16x16x32_f16 (16x32 f16 A-matrix):
//  lane = hi*16 + M ; contiguous 16 halfs per lane hold
//  hi=0: K = 0..15   hi=1: K = 16..31
// ---------------------------------------------------------------------------
__device__ __forceinline__ void a_swizzle(int kk, int& hi, int& idx) {
    hi  = kk >> 4;
    idx = kk & 15;
}

// ---------------------------------------------------------------------------
// K1: modulate + demodulate conv weights, write in A-swizzled f16 layout
// wA[b][mt][tap][kc][lane][16]   (mt = o/16)
// grid: B*CH blocks (b = blk/128, o = blk%128), 128 threads (thread = c)
// ---------------------------------------------------------------------------
__global__ void gen_pack_kernel(const float* __restrict__ cw,   // [O][C][27]
                                const float* __restrict__ s,    // [B][CH]
                                float scale,
                                _Float16* __restrict__ wA) {
    __shared__ float red[CH];
    __shared__ float sig_inv_sh;
    const int b = blockIdx.x / CH;
    const int o = blockIdx.x % CH;
    const int c = threadIdx.x;

    const float sv = s[(size_t)b * CH + c];
    const float* w = cw + ((size_t)o * CH + c) * TAPS;

    float part = 0.0f;
    float g[TAPS];
#pragma unroll
    for (int t = 0; t < TAPS; ++t) {
        float v = scale * w[t] * sv;
        g[t] = v;
        part += v * v;
    }
    red[c] = part;
    __syncthreads();
    for (int off = 64; off > 0; off >>= 1) {
        if (c < off) red[c] += red[c + off];
        __syncthreads();
    }
    if (c == 0) sig_inv_sh = rsqrtf(red[0] + EPS_DEMOD);
    __syncthreads();
    const float sig_inv = sig_inv_sh;

    const int mt = o >> 4;
    const int m  = o & 15;
    const int kc = c >> 5;
    const int kk = c & 31;
    int hi, idx;
    a_swizzle(kk, hi, idx);
    const int lane = hi * 16 + m;
#pragma unroll
    for (int t = 0; t < TAPS; ++t) {
        size_t off = ((((size_t)b * 8 + mt) * TAPS + t) * KC_PER_TAP + kc) * 512
                     + (size_t)lane * 16 + idx;
        wA[off] = (_Float16)(g[t] * sig_inv);
    }
}

// ---------------------------------------------------------------------------
// K2: cast x f32 NCDHW -> f16 channel-last [b][vox][c]
// grid: B*VOX*32 / 256 blocks, 256 threads, 4 channels per thread
// ---------------------------------------------------------------------------
__global__ void gen_cast_kernel(const float* __restrict__ x, _Float16* __restrict__ act) {
    const size_t i  = (size_t)blockIdx.x * blockDim.x + threadIdx.x;
    const int cg = (int)(i & 31);
    const int v  = (int)((i >> 5) & (VOX - 1));
    const int b  = (int)(i >> 20);               // /(32*32768)
    _Float16 h4[4];
#pragma unroll
    for (int j = 0; j < 4; ++j)
        h4[j] = (_Float16)x[((size_t)(b * CH + cg * 4 + j)) * VOX + v];
    *(uint2*)(act + ((size_t)(b * VOX + v)) * CH + cg * 4) = *(const uint2*)h4;
}

// ---------------------------------------------------------------------------
// K3: modulated 3x3x3 conv as implicit GEMM with WMMA f16->f32,
//     fused  + sn*noise + bias, leaky-ReLU.
// One workgroup: batch b, 4x4x4 spatial tile, all 128 output channels.
// 8 waves; wave w owns M-tile w (16 out channels) x 64 voxels (4 N-tiles).
// Halo staged global->LDS via CDNA5 async copies (ASYNCcnt), zero padding
// via plain ds stores (DScnt); both drained before the barrier.
// grid: B*512 blocks of 256 threads
// ---------------------------------------------------------------------------
__global__ void __launch_bounds__(256)
gen_conv_kernel(const _Float16* __restrict__ act_in,   // [b][vox][c] f16
                const _Float16* __restrict__ wA,       // swizzled A
                const float* __restrict__ noise,       // [b][vox]
                const float* __restrict__ sn,          // scalar
                const float* __restrict__ bias,        // [CH]
                _Float16* __restrict__ act_out,        // nullable, [b][vox][c]
                float* __restrict__ f32_out) {         // nullable, [b][c][vox]
    __shared__ _Float16 hs[HPOS * CH];                 // 55296 B, LDS offset 0

    const int blk  = blockIdx.x;
    const int b    = blk >> 9;                         // /512
    const int tile = blk & 511;
    const int d0 = (tile >> 6) * 4;
    const int h0 = ((tile >> 3) & 7) * 4;
    const int w0 = (tile & 7) * 4;

    // ---- halo fill: 216 pos x 256 B, 16 B chunks, async global->LDS ----
    const int tidx = threadIdx.x;
    const char* base_in = (const char*)(act_in + ((size_t)b * VOX) * CH);
#pragma unroll 1
    for (int it = 0; it < 14; ++it) {
        const int idx = it * 256 + tidx;               // 0..3455 (3456 chunks)
        if (idx < HPOS * 16) {
            const int pos = idx >> 4;                  // halo position
            const int cb  = (idx & 15) << 4;           // byte offset in 256B row
            const int pd = pos / 36, ph = (pos % 36) / 6, pw = pos % 6;
            const int gd = d0 + pd - 1, gh = h0 + ph - 1, gw = w0 + pw - 1;
            const unsigned lds_b = (unsigned)(pos * 256 + cb);
            if ((unsigned)gd < 32u && (unsigned)gh < 32u && (unsigned)gw < 32u) {
                const int gv = (gd * 32 + gh) * 32 + gw;
                const void* gp = base_in + (size_t)gv * 256 + cb;
                asm volatile("global_load_async_to_lds_b128 %0, %1, off"
                             :: "v"(lds_b), "v"(gp) : "memory");
            } else {
                uint4 z; z.x = 0u; z.y = 0u; z.z = 0u; z.w = 0u;
                *(uint4*)((char*)hs + pos * 256 + cb) = z;   // zero padding
            }
        }
    }
    asm volatile("s_wait_asynccnt 0x0" ::: "memory");  // drain async copies
    __syncthreads();

    const int wave = tidx >> 5;
    const int lane = tidx & 31;
    const int lnk  = lane & 15;
    const int hi16 = (lane >> 4) * 16;

    // per-N-tile voxel decomposition (tile-local) and halo base offset
    int basep[4], gvx[4];
#pragma unroll
    for (int nt = 0; nt < 4; ++nt) {
        const int n  = nt * 16 + lnk;
        const int vw = n & 3, vh = (n >> 2) & 3, vd = n >> 4;
        basep[nt] = vd * 36 + vh * 6 + vw;
        gvx[nt]   = ((d0 + vd) * 32 + (h0 + vh)) * 32 + (w0 + vw);
    }

    v8f acc0 = {}, acc1 = {}, acc2 = {}, acc3 = {};
    const _Float16* wbase = wA + (((size_t)b * 8 + wave) * TAPS) * KC_PER_TAP * 512;

#pragma unroll 1
    for (int t = 0; t < TAPS; ++t) {
        const int toff = (t / 9) * 36 + ((t / 3) % 3) * 6 + (t % 3);
#pragma unroll
        for (int kc = 0; kc < KC_PER_TAP; ++kc) {
            const v16h a = *(const v16h*)(wbase + ((size_t)t * KC_PER_TAP + kc) * 512
                                          + (size_t)lane * 16);
            const int coff = kc * 32 + hi16;
            v16h b0 = *(const v16h*)(hs + (basep[0] + toff) * CH + coff);
            v16h b1 = *(const v16h*)(hs + (basep[1] + toff) * CH + coff);
            v16h b2 = *(const v16h*)(hs + (basep[2] + toff) * CH + coff);
            v16h b3 = *(const v16h*)(hs + (basep[3] + toff) * CH + coff);
            acc0 = __builtin_amdgcn_wmma_f32_16x16x32_f16(false, a, false, b0, (short)0, acc0, false, false);
            acc1 = __builtin_amdgcn_wmma_f32_16x16x32_f16(false, a, false, b1, (short)0, acc1, false, false);
            acc2 = __builtin_amdgcn_wmma_f32_16x16x32_f16(false, a, false, b2, (short)0, acc2, false, false);
            acc3 = __builtin_amdgcn_wmma_f32_16x16x32_f16(false, a, false, b3, (short)0, acc3, false, false);
        }
    }

    // ---- fused epilogue: + sn*noise + bias, leaky-ReLU ----
    const float snv = *sn;
    const int mrow = (lane >= 16) ? 8 : 0;
    v8f accs[4] = {acc0, acc1, acc2, acc3};
#pragma unroll
    for (int nt = 0; nt < 4; ++nt) {
        const int gv = gvx[nt];
        const float noi = snv * noise[(size_t)b * VOX + gv];
#pragma unroll
        for (int r = 0; r < 8; ++r) {
            const int o = wave * 16 + r + mrow;
            float v = accs[nt][r] + noi + bias[o];
            v = lrelu(v);
            if (act_out) act_out[((size_t)(b * VOX + gv)) * CH + o] = (_Float16)v;
            if (f32_out) f32_out[((size_t)(b * CH + o)) * VOX + gv] = v;
        }
    }
}

// ---------------------------------------------------------------------------
// K4: to_volume 1x1x1 conv (no demod) + bias + lrelu
// grid: B*VOX/256 blocks, 1 thread per voxel
// ---------------------------------------------------------------------------
__global__ void gen_vol_kernel(const _Float16* __restrict__ act,  // [b][vox][c]
                               const float* __restrict__ sv,      // [B][CH]
                               const float* __restrict__ cwv,     // [CH]
                               const float* __restrict__ biasv,   // [1]
                               float* __restrict__ vol) {
    const size_t gid = (size_t)blockIdx.x * blockDim.x + threadIdx.x; // b*VOX+vox
    const int b = (int)(gid >> 15);
    const float scv = 0.08838834764831845f;    // 1/sqrt(128)
    const _Float16* row = act + gid * CH;
    const float* sb = sv + (size_t)b * CH;
    float acc = 0.0f;
#pragma unroll 4
    for (int c = 0; c < CH; ++c)
        acc += (float)row[c] * sb[c] * cwv[c];
    vol[gid] = lrelu(acc * scv + biasv[0]);
}

// ---------------------------------------------------------------------------
extern "C" void kernel_launch(void* const* d_in, const int* in_sizes, int n_in,
                              void* d_out, int out_size, void* d_ws, size_t ws_size,
                              hipStream_t stream) {
    (void)in_sizes; (void)n_in; (void)out_size; (void)ws_size;
    const float* x      = (const float*)d_in[0];
    const float* w_lat  = (const float*)d_in[1];
    const float* noise1 = (const float*)d_in[2];
    const float* noise2 = (const float*)d_in[3];
    const float* W1     = (const float*)d_in[4];
    const float* b1     = (const float*)d_in[5];
    const float* cw1    = (const float*)d_in[6];
    const float* sn1    = (const float*)d_in[7];
    const float* bias1  = (const float*)d_in[8];
    const float* W2     = (const float*)d_in[9];
    const float* b2     = (const float*)d_in[10];
    const float* cw2    = (const float*)d_in[11];
    const float* sn2    = (const float*)d_in[12];
    const float* bias2  = (const float*)d_in[13];
    const float* Wv     = (const float*)d_in[14];
    const float* bv     = (const float*)d_in[15];
    const float* cwv    = (const float*)d_in[16];
    const float* biasv  = (const float*)d_in[17];

    float* h_out   = (float*)d_out;                                  // [8,128,32^3]
    float* vol_out = (float*)d_out + (size_t)BATCH * CH * VOX;       // [8,1,32^3]

    // workspace layout
    char* ws = (char*)d_ws;
    const size_t WA_BYTES  = (size_t)BATCH * 8 * TAPS * KC_PER_TAP * 512 * sizeof(_Float16); // ~7 MB
    const size_t ACT_BYTES = (size_t)BATCH * VOX * CH * sizeof(_Float16);                    // 64 MB
    float*    s_ws  = (float*)ws;                                   // 3*8*128 f32
    _Float16* wA1   = (_Float16*)(ws + 16384);
    _Float16* wA2   = (_Float16*)(ws + 16384 + WA_BYTES);
    _Float16* bufA  = (_Float16*)(ws + 16384 + 2 * WA_BYTES);       // act0, reused as act2
    _Float16* act1  = (_Float16*)(ws + 16384 + 2 * WA_BYTES + ACT_BYTES);

    float* s1 = s_ws;
    float* s2 = s_ws + BATCH * CH;
    float* sv = s_ws + 2 * BATCH * CH;

    const float scale_conv = 1.0f / sqrtf((float)(CH * TAPS));       // 1/sqrt(3456)

    // styles (s1, s2, sv)
    gen_styles_kernel<<<3 * BATCH, CH, 0, stream>>>(w_lat, W1, b1, W2, b2, Wv, bv, s_ws);
    // x f32 -> f16 channel-last
    gen_cast_kernel<<<(BATCH * VOX * 32) / 256, 256, 0, stream>>>(x, bufA);
    // modulated + demodulated weights, A-swizzled
    gen_pack_kernel<<<BATCH * CH, CH, 0, stream>>>(cw1, s1, scale_conv, wA1);
    gen_pack_kernel<<<BATCH * CH, CH, 0, stream>>>(cw2, s2, scale_conv, wA2);
    // StyleBlock 1: conv(x)->act1 (f16 only)
    gen_conv_kernel<<<BATCH * 512, 256, 0, stream>>>(bufA, wA1, noise1, sn1, bias1,
                                                     act1, nullptr);
    // StyleBlock 2: conv(act1)->h (f32 out) + act2 f16 (reuses bufA)
    gen_conv_kernel<<<BATCH * 512, 256, 0, stream>>>(act1, wA2, noise2, sn2, bias2,
                                                     bufA, h_out);
    // ToVolume
    gen_vol_kernel<<<(BATCH * VOX) / 256, 256, 0, stream>>>(bufA, sv, cwv, biasv, vol_out);
}
